// EdgeMessage_60138132079231
// MI455X (gfx1250) — matrix operand order repaired and checked by reading, not compiled, not run on hardware
//
#include <hip/hip_runtime.h>

#define E_TOT 100000
#define VDIM 64
#define KDIM 128
#define MDIM 64
#define TE   64            // edges per workgroup (4 WMMA M-tiles)
#define ELDS_STRIDE 136    // bf16 per edge row in LDS (128 + 8 pad -> conflict-free b128 reads)
#define VLDS_STRIDE 68     // f32 per transposed vertex row (64 + 4 pad -> 16B aligned, conflict-free)

typedef __attribute__((ext_vector_type(16))) __bf16 v16bf;
typedef __attribute__((ext_vector_type(8)))  float  v8f;

union BfPack { uint4 q[2]; v16bf v; };

// one-time f32 -> bf16 (RNE) conversion of W into workspace (1 MB, stays in L2)
__global__ void cvt_w_bf16(const float* __restrict__ W, unsigned short* __restrict__ out, int n) {
    int i = blockIdx.x * 256 + threadIdx.x;
    if (i < n) {
        unsigned u = __float_as_uint(W[i]);
        u += 0x7FFFu + ((u >> 16) & 1u);
        out[i] = (unsigned short)(u >> 16);
    }
}

__global__ __launch_bounds__(256)
void edge_message_kernel(const float* __restrict__ vertices,      // [E,64]
                         const float* __restrict__ edges,         // [E,128]
                         const unsigned short* __restrict__ Wbf,  // [4096,128] bf16
                         const float* __restrict__ bias,          // [4096]
                         float* __restrict__ out)                 // [E,64]
{
    __shared__ __align__(16) unsigned short eLds[TE * ELDS_STRIDE];   // edges tile, bf16
    __shared__ __align__(16) float          vLds[VDIM * VLDS_STRIDE]; // vertices tile, transposed [v][e]

    const int tid   = threadIdx.x;
    const int lane  = tid & 31;
    const int wave  = tid >> 5;
    const int h     = lane >> 4;   // half-wave (K/M-half selector per WMMA layout)
    const int lcol  = lane & 15;   // N column / M row index
    const int ebase = blockIdx.x * TE;

    // ---- stage edges tile f32 -> bf16 into LDS (vectorized, coalesced) ----
    #pragma unroll
    for (int i = 0; i < (TE * KDIM) / (256 * 4); ++i) {     // 8 iters of float4
        int idx = tid + i * 256;                            // over 2048 float4
        int er  = idx >> 5;                                 // 0..63
        int k4  = (idx & 31) * 4;                           // 0,4,...,124
        int eg  = ebase + er;
        float4 f = (eg < E_TOT) ? ((const float4*)(edges + eg * KDIM))[k4 >> 2]
                                : make_float4(0.f, 0.f, 0.f, 0.f);
        unsigned short s[4];
        float ff[4] = {f.x, f.y, f.z, f.w};
        #pragma unroll
        for (int c = 0; c < 4; ++c) {
            unsigned u = __float_as_uint(ff[c]);
            u += 0x7FFFu + ((u >> 16) & 1u);
            s[c] = (unsigned short)(u >> 16);
        }
        *(ushort4*)&eLds[er * ELDS_STRIDE + k4] = make_ushort4(s[0], s[1], s[2], s[3]);
    }
    // ---- stage vertices tile transposed into LDS ----
    #pragma unroll
    for (int i = 0; i < (TE * VDIM) / (256 * 4); ++i) {     // 4 iters of float4
        int idx = tid + i * 256;                            // over 1024 float4
        int er  = idx >> 4;                                 // 0..63
        int v4  = (idx & 15) * 4;                           // 0,4,...,60
        int eg  = ebase + er;
        float4 f = (eg < E_TOT) ? ((const float4*)(vertices + eg * VDIM))[v4 >> 2]
                                : make_float4(0.f, 0.f, 0.f, 0.f);
        vLds[(v4 + 0) * VLDS_STRIDE + er] = f.x;
        vLds[(v4 + 1) * VLDS_STRIDE + er] = f.y;
        vLds[(v4 + 2) * VLDS_STRIDE + er] = f.z;
        vLds[(v4 + 3) * VLDS_STRIDE + er] = f.w;
    }
    __syncthreads();

    // ---- hoist A (edges) tiles into WMMA 16-bit A-layout registers ----
    // lane: M = lcol (+16*mt); elements 0..7 -> K = ks*32 + h*8 + i, 8..15 -> +16
    v16bf A[4][4];
    #pragma unroll
    for (int mt = 0; mt < 4; ++mt) {
        #pragma unroll
        for (int ks = 0; ks < 4; ++ks) {
            int row = mt * 16 + lcol;
            int kb  = ks * 32 + h * 8;
            BfPack t;
            t.q[0] = *(const uint4*)&eLds[row * ELDS_STRIDE + kb];
            t.q[1] = *(const uint4*)&eLds[row * ELDS_STRIDE + kb + 16];
            A[mt][ks] = t.v;
        }
    }

    // ---- each wave computes 8 m-values over the 64-edge tile ----
    for (int mi = 0; mi < 8; ++mi) {
        int m = wave * 8 + mi;
        float acc[4][8];
        #pragma unroll
        for (int mt = 0; mt < 4; ++mt)
            #pragma unroll
            for (int j = 0; j < 8; ++j) acc[mt][j] = 0.0f;

        for (int vs = 0; vs < 4; ++vs) {
            int nrow = m * 64 + vs * 16 + lcol;   // W row handled by this lane (N column)
            float bval = bias[nrow];              // bias folds into C init
            // B tiles: lane holds column N = lcol, elements 0..7 -> K = ks*32 + h*8 + i, 8..15 -> +16
            v16bf Bm[4];
            #pragma unroll
            for (int ks = 0; ks < 4; ++ks) {
                int k0 = ks * 32 + h * 8;
                const uint4* q = (const uint4*)&Wbf[nrow * KDIM + k0];
                BfPack t;
                t.q[0] = q[0];
                t.q[1] = q[2];
                Bm[ks] = t.v;
            }
            int vcol = vs * 16 + lcol;
            #pragma unroll
            for (int mt = 0; mt < 4; ++mt) {
                v8f c;
                #pragma unroll
                for (int j = 0; j < 8; ++j) c[j] = bval;   // proj = edges@W^T + b
                #pragma unroll
                for (int ks = 0; ks < 4; ++ks) {
                    c = __builtin_amdgcn_wmma_f32_16x16x32_bf16(
                            false, A[mt][ks], false, Bm[ks], (short)0, c, false, false);
                }
                // vertices tile in exact C layout: element (M=j+8h, N=lcol) = vert[e, v]
                const float4* vp = (const float4*)&vLds[vcol * VLDS_STRIDE + mt * 16 + h * 8];
                float4 va0 = vp[0], va1 = vp[1];
                acc[mt][0] += c[0] * va0.x;  acc[mt][1] += c[1] * va0.y;
                acc[mt][2] += c[2] * va0.z;  acc[mt][3] += c[3] * va0.w;
                acc[mt][4] += c[4] * va1.x;  acc[mt][5] += c[5] * va1.y;
                acc[mt][6] += c[6] * va1.z;  acc[mt][7] += c[7] * va1.w;
            }
        }

        // reduce over the 16-lane v dimension (xor masks stay within 16-lane halves) and store
        #pragma unroll
        for (int mt = 0; mt < 4; ++mt) {
            #pragma unroll
            for (int j = 0; j < 8; ++j) {
                float s = acc[mt][j];
                s += __shfl_xor(s, 1, 32);
                s += __shfl_xor(s, 2, 32);
                s += __shfl_xor(s, 4, 32);
                s += __shfl_xor(s, 8, 32);
                acc[mt][j] = s;
            }
            if (lcol == 0) {
                int e0 = ebase + mt * 16 + h * 8;
                #pragma unroll
                for (int j = 0; j < 8; ++j) {
                    int e = e0 + j;
                    if (e < E_TOT) out[e * MDIM + m] = acc[mt][j];
                }
            }
        }
    }
}

extern "C" void kernel_launch(void* const* d_in, const int* in_sizes, int n_in,
                              void* d_out, int out_size, void* d_ws, size_t ws_size,
                              hipStream_t stream) {
    (void)in_sizes; (void)n_in; (void)out_size; (void)ws_size;
    const float* vertices = (const float*)d_in[0];
    const float* edges    = (const float*)d_in[1];
    const float* W        = (const float*)d_in[2];   // [4096,128]
    const float* b        = (const float*)d_in[3];   // [4096]
    float* out            = (float*)d_out;           // [E,64]

    unsigned short* Wbf = (unsigned short*)d_ws;     // 4096*128*2 = 1 MB
    const int nW = (MDIM * VDIM) * KDIM;             // 524288
    cvt_w_bf16<<<nW / 256, 256, 0, stream>>>(W, Wbf, nW);

    const int grid = (E_TOT + TE - 1) / TE;          // 1563
    edge_message_kernel<<<grid, 256, 0, stream>>>(vertices, edges, Wbf, b, out);
}
